// CoordOffsetAdapter_919123001514
// MI455X (gfx1250) — matrix operand order repaired and checked by reading, not compiled, not run on hardware
//
#include <hip/hip_runtime.h>

#define VOCAB    152704
#define DIM      1024
#define TOKENS   2048
#define COORD_LO 151670
#define COORD_HI 152669

typedef __attribute__((ext_vector_type(16))) _Float16       v16h;
typedef __attribute__((ext_vector_type(4)))  _Float16       h4;
typedef __attribute__((ext_vector_type(2)))  __fp16         g2;   // cvt_pkrtz result type
typedef __attribute__((ext_vector_type(8)))  float          v8f;
typedef __attribute__((ext_vector_type(4)))  float          f32x4;
typedef __attribute__((ext_vector_type(4)))  unsigned int   u32x4;

union Frag { u32x4 q[2]; v16h v; };
union H4u  { h4 v; g2 g[2]; };      // __fp16 and _Float16 share the same bits

// Split fp32 -> fp16 hi + fp16 lo (residual). hi via v_cvt_pk_rtz_f16_f32;
// residual is exact in fp32, then packed the same way. hi+lo carries ~22 bits.
__device__ __forceinline__ void split4(const f32x4 x, h4& hi, h4& lo) {
  g2 h01 = __builtin_amdgcn_cvt_pkrtz(x[0], x[1]);
  g2 h23 = __builtin_amdgcn_cvt_pkrtz(x[2], x[3]);
  g2 l01 = __builtin_amdgcn_cvt_pkrtz(x[0] - (float)h01[0], x[1] - (float)h01[1]);
  g2 l23 = __builtin_amdgcn_cvt_pkrtz(x[2] - (float)h23[0], x[3] - (float)h23[1]);
  H4u H, L;
  H.g[0] = h01; H.g[1] = h23;
  L.g[0] = l01; L.g[1] = l23;
  hi = H.v; lo = L.v;
}

// ---------------- embedding gather + masked offset add ----------------
__global__ void embed_kernel(const int* __restrict__ ids,
                             const float* __restrict__ table,
                             const float* __restrict__ off,
                             float* __restrict__ emb) {
  const int t = blockIdx.x;
  const int c = threadIdx.x * 4;
  const int id = ids[t];
  f32x4 v = *(const f32x4*)(table + (size_t)id * DIM + c);
  if (id >= COORD_LO && id <= COORD_HI) {
    f32x4 e = *(const f32x4*)(off + (size_t)(id - COORD_LO) * DIM + c);
    v = v + e;
  }
  *(f32x4*)(emb + (size_t)t * DIM + c) = v;
}

// ---------------- fused lm_head GEMM (fp16 hi/lo split, fp32 accum) ----------------
// WG tile: 256(M) x 128(N), K-chunks of 32, double-buffered LDS.
// 8 waves as 4(M) x 2(N); each wave: 64x64 macro-tile = 4x4 subtiles of 16x16.
#define MT 256
#define NT 128
#define AP 40    // 32 K-halves + 8 pad   (80B rows, 16B aligned)
#define BP 136   // 128 N-halves + 8 pad  (272B rows, 16B aligned)
#define A_BUF (MT * AP)          // halves per buffer
#define B_BUF (32 * BP)
#define A_HI_OFF 0
#define A_LO_OFF (2 * A_BUF)
#define B_HI_OFF (A_LO_OFF + 2 * A_BUF)
#define B_LO_OFF (B_HI_OFF + 2 * B_BUF)
#define SMEM_HALVES (B_LO_OFF + 2 * B_BUF)   // 58368 halves = 116736 B

__global__ void __launch_bounds__(256)
lmhead_gemm(const float* __restrict__ H, const float* __restrict__ W,
            const float* __restrict__ E, float* __restrict__ out) {
  extern __shared__ __align__(16) _Float16 smem[];
  _Float16* aHs = smem + A_HI_OFF;
  _Float16* aLs = smem + A_LO_OFF;
  _Float16* bHs = smem + B_HI_OFF;
  _Float16* bLs = smem + B_LO_OFF;

  const int tid  = threadIdx.x;
  const int m0   = (blockIdx.x & 7) * MT;    // M fastest -> W N-band reused in L2
  const int n0   = (blockIdx.x >> 3) * NT;
  const int lane = tid & 31;
  const int wave = tid >> 5;
  const int wm   = wave & 3;    // 4 M-groups of 64 rows
  const int wn   = wave >> 2;   // 2 N-groups of 64 cols

  v8f acc[4][4];
#pragma unroll
  for (int i = 0; i < 4; ++i)
#pragma unroll
    for (int j = 0; j < 4; ++j)
      acc[i][j] = (v8f){0.f, 0.f, 0.f, 0.f, 0.f, 0.f, 0.f, 0.f};

  const int lrow = tid >> 3;        // 0..31: row/col within a 32-wide pass
  const int lk   = (tid & 7) * 4;   // 0..28: float4 along K

  f32x4 av[8], bv[4];

  // ---- prologue: global-load chunk 0 ----
#pragma unroll
  for (int p = 0; p < 8; ++p)
    av[p] = *(const f32x4*)(H + (size_t)(m0 + lrow + p * 32) * DIM + lk);
#pragma unroll
  for (int p = 0; p < 4; ++p) {
    const int cg = n0 + lrow + p * 32;
    f32x4 x = *(const f32x4*)(W + (size_t)cg * DIM + lk);
    if (cg >= COORD_LO && cg <= COORD_HI)   // fold coord scatter-add into weights
      x = x + *(const f32x4*)(E + (size_t)(cg - COORD_LO) * DIM + lk);
    bv[p] = x;
  }
  // ---- split + store chunk 0 into buffer 0 ----
#pragma unroll
  for (int p = 0; p < 8; ++p) {
    h4 h, l; split4(av[p], h, l);
    *(h4*)(aHs + (lrow + p * 32) * AP + lk) = h;
    *(h4*)(aLs + (lrow + p * 32) * AP + lk) = l;
  }
#pragma unroll
  for (int p = 0; p < 4; ++p) {
    h4 h, l; split4(bv[p], h, l);
#pragma unroll
    for (int e = 0; e < 4; ++e) {
      bHs[(lk + e) * BP + lrow + p * 32] = h[e];
      bLs[(lk + e) * BP + lrow + p * 32] = l[e];
    }
  }
  __syncthreads();

  int buf = 0;
  for (int kt = 0; kt < 32; ++kt) {
    // ---- issue next chunk's global loads (overlap with WMMA below) ----
    if (kt < 31) {
      const int kb = (kt + 1) * 32;
#pragma unroll
      for (int p = 0; p < 8; ++p)
        av[p] = *(const f32x4*)(H + (size_t)(m0 + lrow + p * 32) * DIM + kb + lk);
#pragma unroll
      for (int p = 0; p < 4; ++p) {
        const int cg = n0 + lrow + p * 32;
        f32x4 x = *(const f32x4*)(W + (size_t)cg * DIM + kb + lk);
        if (cg >= COORD_LO && cg <= COORD_HI)
          x = x + *(const f32x4*)(E + (size_t)(cg - COORD_LO) * DIM + kb + lk);
        bv[p] = x;
      }
    }

    // ---- compute: 4 Msub x 4 Nsub x 3 split-WMMAs from LDS buf ----
    const _Float16* aHb = aHs + buf * A_BUF;
    const _Float16* aLb = aLs + buf * A_BUF;
    const _Float16* bHb = bHs + buf * B_BUF;
    const _Float16* bLb = bLs + buf * B_BUF;
#pragma unroll
    for (int ms = 0; ms < 4; ++ms) {
      const int aoff = (wm * 64 + ms * 16 + (lane & 15)) * AP + ((lane >> 4) * 8);
      Frag fah, fal;
      fah.q[0] = *(const u32x4*)(aHb + aoff);
      fah.q[1] = *(const u32x4*)(aHb + aoff + 16);
      fal.q[0] = *(const u32x4*)(aLb + aoff);
      fal.q[1] = *(const u32x4*)(aLb + aoff + 16);
#pragma unroll
      for (int ns = 0; ns < 4; ++ns) {
        const int boff = ((lane >> 4) * 8) * BP + wn * 64 + ns * 16 + (lane & 15);
        Frag fbh, fbl;
#pragma unroll
        for (int s = 0; s < 16; ++s) {
          const int kk = (s & 7) + ((s >> 3) << 4);
          fbh.v[s] = bHb[boff + kk * BP];
          fbl.v[s] = bLb[boff + kk * BP];
        }
        acc[ms][ns] = __builtin_amdgcn_wmma_f32_16x16x32_f16(
            false, fah.v, false, fbh.v, (short)0, acc[ms][ns], false, false);
        acc[ms][ns] = __builtin_amdgcn_wmma_f32_16x16x32_f16(
            false, fah.v, false, fbl.v, (short)0, acc[ms][ns], false, false);
        acc[ms][ns] = __builtin_amdgcn_wmma_f32_16x16x32_f16(
            false, fal.v, false, fbh.v, (short)0, acc[ms][ns], false, false);
      }
    }

    // ---- split + store next chunk into other buffer ----
    if (kt < 31) {
      _Float16* aHn = aHs + (buf ^ 1) * A_BUF;
      _Float16* aLn = aLs + (buf ^ 1) * A_BUF;
      _Float16* bHn = bHs + (buf ^ 1) * B_BUF;
      _Float16* bLn = bLs + (buf ^ 1) * B_BUF;
#pragma unroll
      for (int p = 0; p < 8; ++p) {
        h4 h, l; split4(av[p], h, l);
        *(h4*)(aHn + (lrow + p * 32) * AP + lk) = h;
        *(h4*)(aLn + (lrow + p * 32) * AP + lk) = l;
      }
#pragma unroll
      for (int p = 0; p < 4; ++p) {
        h4 h, l; split4(bv[p], h, l);
#pragma unroll
        for (int e = 0; e < 4; ++e) {
          bHn[(lk + e) * BP + lrow + p * 32] = h[e];
          bLn[(lk + e) * BP + lrow + p * 32] = l[e];
        }
      }
    }
    __syncthreads();
    buf ^= 1;
  }

  // ---- epilogue: C/D layout (lane<16: M=j, lane>=16: M=8+j; N=lane%16) ----
#pragma unroll
  for (int ms = 0; ms < 4; ++ms) {
    const int r0 = m0 + wm * 64 + ms * 16 + ((lane >> 4) << 3);
#pragma unroll
    for (int ns = 0; ns < 4; ++ns) {
      const int c = n0 + wn * 64 + ns * 16 + (lane & 15);
#pragma unroll
      for (int j = 0; j < 8; ++j)
        out[(size_t)(r0 + j) * VOCAB + c] = acc[ms][ns][j];
    }
  }
}

extern "C" void kernel_launch(void* const* d_in, const int* in_sizes, int n_in,
                              void* d_out, int out_size, void* d_ws, size_t ws_size,
                              hipStream_t stream) {
  const int*   ids    = (const int*)d_in[0];
  const float* hidden = (const float*)d_in[1];
  const float* table  = (const float*)d_in[2];
  const float* headW  = (const float*)d_in[3];
  const float* offs   = (const float*)d_in[4];
  float* out    = (float*)d_out;
  float* emb    = out;                              // [2048,1024]
  float* logits = out + (size_t)TOKENS * DIM;       // [2048,152704]

  embed_kernel<<<TOKENS, 256, 0, stream>>>(ids, table, offs, emb);

  const int grid = (TOKENS / MT) * (VOCAB / NT);    // 8 M-tiles x 1193 N-tiles
  const size_t smem_bytes = (size_t)SMEM_HALVES * sizeof(_Float16); // ~114 KB
  lmhead_gemm<<<grid, 256, smem_bytes, stream>>>(hidden, headW, offs, logits);
}
